// CalculateAttention_12979391169218
// MI455X (gfx1250) — matrix operand order repaired and checked by reading, not compiled, not run on hardware
//
#include <hip/hip_runtime.h>
#include <math.h>

typedef __attribute__((ext_vector_type(16))) __bf16 v16bf;
typedef __attribute__((ext_vector_type(8)))  __bf16 v8bf;
typedef __attribute__((ext_vector_type(8)))  float  v8f;

#define S_LEN 2048
#define HDIM  64
#define QT    128   // queries per workgroup (8 waves x 16)
#define KT    64    // keys per iteration (two 32-key WMMA k-groups)

__global__ __launch_bounds__(256)
void fattn_wmma_kernel(const float* __restrict__ Qg, const float* __restrict__ Kg,
                       const float* __restrict__ Vg, float* __restrict__ Og) {
    __shared__ __align__(32) __bf16 sK [2][KT * HDIM];   // [buf][key][d]   16 KB
    __shared__ __align__(32) __bf16 sVt[2][HDIM * KT];   // [buf][d][key]   16 KB

    const int tid  = threadIdx.x;
    const int lane = tid & 31;
    const int wave = tid >> 5;
    const int half = lane >> 4;   // 0: lanes 0-15, 1: lanes 16-31
    const int ln   = lane & 15;

    const int bh = blockIdx.x >> 4;                 // head index (B*H = 32)
    const int q0 = (blockIdx.x & 15) * QT + wave * 16;

    const size_t hb = (size_t)bh * S_LEN * HDIM;
    const float* Qh = Qg + hb;
    const float* Kh = Kg + hb;
    const float* Vh = Vg + hb;
    float*       Oh = Og + hb;

    // fold softmax scale AND log2(e) into Q: scores in log2 domain,
    // softmax is p = exp2(st - m) with raw v_exp_f32.
    const float qscale = 0.125f * 1.44269504088896340736f;

    // ---- Q as B-fragments of Q^T (32x16 per d-step): d = s*32 + 16*(e>>3) + 8*half + (e&7), N = ln ----
    v16bf qb[2];
    #pragma unroll
    for (int s = 0; s < 2; ++s) {
        const float* src = Qh + (size_t)(q0 + ln) * HDIM + s * 32 + half * 8;
        const float4 f0 = ((const float4*)src)[0];
        const float4 f1 = ((const float4*)src)[1];
        const float4 f2 = ((const float4*)src)[4];
        const float4 f3 = ((const float4*)src)[5];
        qb[s][ 0] = (__bf16)(f0.x * qscale); qb[s][ 1] = (__bf16)(f0.y * qscale);
        qb[s][ 2] = (__bf16)(f0.z * qscale); qb[s][ 3] = (__bf16)(f0.w * qscale);
        qb[s][ 4] = (__bf16)(f1.x * qscale); qb[s][ 5] = (__bf16)(f1.y * qscale);
        qb[s][ 6] = (__bf16)(f1.z * qscale); qb[s][ 7] = (__bf16)(f1.w * qscale);
        qb[s][ 8] = (__bf16)(f2.x * qscale); qb[s][ 9] = (__bf16)(f2.y * qscale);
        qb[s][10] = (__bf16)(f2.z * qscale); qb[s][11] = (__bf16)(f2.w * qscale);
        qb[s][12] = (__bf16)(f3.x * qscale); qb[s][13] = (__bf16)(f3.y * qscale);
        qb[s][14] = (__bf16)(f3.z * qscale); qb[s][15] = (__bf16)(f3.w * qscale);
    }

    // O^T accumulators: tile nt, reg r = dim nt*16 + r + 8*half, lane = query q0+ln
    v8f acc[4];
    #pragma unroll
    for (int nt = 0; nt < 4; ++nt) acc[nt] = (v8f){0.f,0.f,0.f,0.f,0.f,0.f,0.f,0.f};
    float mrow = -INFINITY;   // per-lane running max (log2 domain)
    float lrow = 0.f;         // per-lane running sum

    // A-row remap for S^T: group c, tile t, row M -> key = 32c + 8t + (M&7) + 16*(M>>3)
    // => lane(h,ln) reg r holds key 32c + 8t + r + 16h == B-element e = 8t + r of P^T.
    const int rowK = ((ln & 7) + 16 * (ln >> 3)) * HDIM;

    // staging geometry: one contiguous 64x64 tile, 4096 floats, 16/thread
    const int base = tid * 16;       // stays within one 64-wide row
    const int skey = base >> 6;
    const int d0   = base & 63;

    float4 kreg[4], vreg[4];
    {
        const float* kp = Kh + base;
        const float* vp = Vh + base;
        #pragma unroll
        for (int j = 0; j < 4; ++j) {
            kreg[j] = ((const float4*)kp)[j];
            vreg[j] = ((const float4*)vp)[j];
        }
    }
    // store tile 0 into buffer 0
    {
        v8bf ka, kb2;
        ka[0]=(__bf16)kreg[0].x; ka[1]=(__bf16)kreg[0].y; ka[2]=(__bf16)kreg[0].z; ka[3]=(__bf16)kreg[0].w;
        ka[4]=(__bf16)kreg[1].x; ka[5]=(__bf16)kreg[1].y; ka[6]=(__bf16)kreg[1].z; ka[7]=(__bf16)kreg[1].w;
        kb2[0]=(__bf16)kreg[2].x; kb2[1]=(__bf16)kreg[2].y; kb2[2]=(__bf16)kreg[2].z; kb2[3]=(__bf16)kreg[2].w;
        kb2[4]=(__bf16)kreg[3].x; kb2[5]=(__bf16)kreg[3].y; kb2[6]=(__bf16)kreg[3].z; kb2[7]=(__bf16)kreg[3].w;
        *(v8bf*)&sK[0][base]     = ka;
        *(v8bf*)&sK[0][base + 8] = kb2;
        #pragma unroll
        for (int j = 0; j < 4; ++j) {
            sVt[0][(d0 + 4*j + 0)*KT + skey] = (__bf16)vreg[j].x;
            sVt[0][(d0 + 4*j + 1)*KT + skey] = (__bf16)vreg[j].y;
            sVt[0][(d0 + 4*j + 2)*KT + skey] = (__bf16)vreg[j].z;
            sVt[0][(d0 + 4*j + 3)*KT + skey] = (__bf16)vreg[j].w;
        }
    }

    for (int kv0 = 0; kv0 < S_LEN; kv0 += KT) {
        const int cur = (kv0 >> 6) & 1;
        const int nxt = cur ^ 1;
        // branch-free: always prefetch the (wrapped) next tile; last iteration
        // redundantly re-stages tile 0 into the dead buffer (no race, no reader).
        const int nk  = (kv0 + KT) & (S_LEN - 1);
        __syncthreads();   // buf[cur] stores visible; buf[nxt] reads (iter-1) done

        // issue next tile's global loads now; latency hidden behind compute below
        {
            const float* kp = Kh + (size_t)nk * HDIM + base;
            const float* vp = Vh + (size_t)nk * HDIM + base;
            #pragma unroll
            for (int j = 0; j < 4; ++j) {
                kreg[j] = ((const float4*)kp)[j];
                vreg[j] = ((const float4*)vp)[j];
            }
        }

        // ---- S^T = K * Q^T, 4 remapped 16-key tiles (2 groups x 2 tiles) ----
        v8f st[2][2];
        #pragma unroll
        for (int c = 0; c < 2; ++c)
            #pragma unroll
            for (int t = 0; t < 2; ++t) {
                v8f sc = (v8f){0.f,0.f,0.f,0.f,0.f,0.f,0.f,0.f};
                #pragma unroll
                for (int s = 0; s < 2; ++s) {
                    const __bf16* p0 = &sK[cur][rowK + (32*c + 8*t)*HDIM + s*32 + half*8];
                    const v8bf lo = *(const v8bf*)p0;
                    const v8bf hi = *(const v8bf*)(p0 + 16);
                    const v16bf a = __builtin_shufflevector(lo, hi,
                        0,1,2,3,4,5,6,7,8,9,10,11,12,13,14,15);
                    sc = __builtin_amdgcn_wmma_f32_16x16x32_bf16(false, a, false, qb[s],
                                                                 (short)0, sc, false, false);
                }
                st[c][t] = sc;
            }
        // lane(h,ln): query = ln, st[c][t][r] = score(key 32c + 8t + r + 16h), log2 units

        // ---- online softmax over 64 keys, per-lane ----
        float mx = st[0][0][0];
        #pragma unroll
        for (int c = 0; c < 2; ++c)
            #pragma unroll
            for (int t = 0; t < 2; ++t)
                #pragma unroll
                for (int r = 0; r < 8; ++r)
                    mx = fmaxf(mx, st[c][t][r]);
        mx = fmaxf(mx, __shfl_xor(mx, 16));
        const float mnew  = fmaxf(mrow, mx);
        const float alpha = __builtin_amdgcn_exp2f(mrow - mnew);   // raw v_exp_f32

        v16bf pb[2];
        float ssum = 0.f;
        #pragma unroll
        for (int c = 0; c < 2; ++c)
            #pragma unroll
            for (int t = 0; t < 2; ++t)
                #pragma unroll
                for (int r = 0; r < 8; ++r) {
                    const float e = __builtin_amdgcn_exp2f(st[c][t][r] - mnew);
                    ssum += e;
                    pb[c][t*8 + r] = (__bf16)e;   // B-element e = 8t + r, group c
                }
        ssum += __shfl_xor(ssum, 16);
        lrow = lrow * alpha + ssum;
        mrow = mnew;

        // ---- rescale O^T accumulators: pure per-lane scalar ----
        #pragma unroll
        for (int nt = 0; nt < 4; ++nt)
            #pragma unroll
            for (int r = 0; r < 8; ++r)
                acc[nt][r] *= alpha;

        // ---- O^T += V^T * P^T : two 32-key k-steps per dim tile ----
        // A 16x32 (V^T): M = dim nt*16 + ln, K(e) = 32c + 16*half + e
        #pragma unroll
        for (int nt = 0; nt < 4; ++nt)
            #pragma unroll
            for (int c = 0; c < 2; ++c) {
                const __bf16* p0 = &sVt[cur][(nt*16 + ln) * KT + 32*c + half*16];
                const v8bf lo = *(const v8bf*)p0;
                const v8bf hi = *(const v8bf*)(p0 + 8);
                const v16bf a = __builtin_shufflevector(lo, hi,
                    0,1,2,3,4,5,6,7,8,9,10,11,12,13,14,15);
                acc[nt] = __builtin_amdgcn_wmma_f32_16x16x32_bf16(false, a, false, pb[c],
                                                                  (short)0, acc[nt], false, false);
            }

        // ---- convert + store next tile into the other buffer (always) ----
        {
            v8bf ka, kb2;
            ka[0]=(__bf16)kreg[0].x; ka[1]=(__bf16)kreg[0].y; ka[2]=(__bf16)kreg[0].z; ka[3]=(__bf16)kreg[0].w;
            ka[4]=(__bf16)kreg[1].x; ka[5]=(__bf16)kreg[1].y; ka[6]=(__bf16)kreg[1].z; ka[7]=(__bf16)kreg[1].w;
            kb2[0]=(__bf16)kreg[2].x; kb2[1]=(__bf16)kreg[2].y; kb2[2]=(__bf16)kreg[2].z; kb2[3]=(__bf16)kreg[2].w;
            kb2[4]=(__bf16)kreg[3].x; kb2[5]=(__bf16)kreg[3].y; kb2[6]=(__bf16)kreg[3].z; kb2[7]=(__bf16)kreg[3].w;
            *(v8bf*)&sK[nxt][base]     = ka;
            *(v8bf*)&sK[nxt][base + 8] = kb2;
            #pragma unroll
            for (int j = 0; j < 4; ++j) {
                sVt[nxt][(d0 + 4*j + 0)*KT + skey] = (__bf16)vreg[j].x;
                sVt[nxt][(d0 + 4*j + 1)*KT + skey] = (__bf16)vreg[j].y;
                sVt[nxt][(d0 + 4*j + 2)*KT + skey] = (__bf16)vreg[j].z;
                sVt[nxt][(d0 + 4*j + 3)*KT + skey] = (__bf16)vreg[j].w;
            }
        }
    }

    // ---- epilogue: per-lane normalize, store O^T layout (scattered, once) ----
    const float inv = 1.0f / lrow;
    const int   row = q0 + ln;
    #pragma unroll
    for (int nt = 0; nt < 4; ++nt)
        #pragma unroll
        for (int r = 0; r < 8; ++r)
            Oh[(size_t)row * HDIM + nt*16 + r + 8*half] = acc[nt][r] * inv;
}

extern "C" void kernel_launch(void* const* d_in, const int* in_sizes, int n_in,
                              void* d_out, int out_size, void* d_ws, size_t ws_size,
                              hipStream_t stream) {
    const float* Q = (const float*)d_in[0];
    const float* K = (const float*)d_in[1];
    const float* V = (const float*)d_in[2];
    float*       O = (float*)d_out;
    // grid: (B*H = 32) heads x (S/QT = 16) query blocks
    dim3 grid(32 * (S_LEN / QT)), block(256);
    fattn_wmma_kernel<<<grid, block, 0, stream>>>(Q, K, V, O);
}
